// DispersionD3_32469952757807
// MI455X (gfx1250) — compile-verified, matrix operand order checked by reading, not run on hardware
//
#include <hip/hip_runtime.h>

typedef __attribute__((ext_vector_type(16))) _Float16 v16h;
typedef __attribute__((ext_vector_type(8)))  float    v8f;

#define NELEM   7
#define NCOMBO  49
#define NREF2   25
#define TABSTR  32                    // padded per-combo stride (shift addressing)
#define K1c     16.0f
#define K2c     (4.0f / 3.0f)
#define S6c     1.0f
#define S8c     0.2641f
#define A1c     0.3f
#define A2c     5.0f
// -K3 * log2(e), -K1 * log2(e): fold exp(x) = exp2(x*log2e) into one constant
#define NEGK3L2E  (-5.770780163555854f)
#define NEGK1L2E  (-23.083120654223414f)

__device__ __forceinline__ void atomic_fadd(float* p, float v) {
  __hip_atomic_fetch_add(p, v, __ATOMIC_RELAXED, __HIP_MEMORY_SCOPE_AGENT);
}

__device__ __forceinline__ float fast_rcp(float x)  { return __builtin_amdgcn_rcpf(x); }
__device__ __forceinline__ float fast_sqrt(float x) { return __builtin_amdgcn_sqrtf(x); }
__device__ __forceinline__ float fast_exp2(float x) { return __builtin_amdgcn_exp2f(x); }

// static 8-way select (7 v_cndmask, select bits loop-invariant)
__device__ __forceinline__ float sel8(const v8f& v, bool b0, bool b1, bool b2) {
  float t0 = b0 ? v[1] : v[0];
  float t1 = b0 ? v[3] : v[2];
  float t2 = b0 ? v[5] : v[4];
  float t3 = b0 ? v[7] : v[6];
  float u0 = b1 ? t1 : t0;
  float u1 = b1 ? t3 : t2;
  return b2 ? u1 : u0;
}

// ---------------- zero workspace ----------------
__global__ void zero_ws_kernel(float* __restrict__ ws, int n) {
  int t = blockIdx.x * blockDim.x + threadIdx.x;
  if (t < n) ws[t] = 0.0f;
}

// ---------------- pass 1: coordination numbers (scatter-add) ----------------
__global__ __launch_bounds__(256)
void cn_pass_kernel(const int* __restrict__ idx, const int* __restrict__ species,
                    const float* __restrict__ dist, const float* __restrict__ cr,
                    float* __restrict__ cn, int P) {
  __shared__ float s_cr[NELEM];
  if (threadIdx.x < NELEM) s_cr[threadIdx.x] = cr[threadIdx.x];
  __syncthreads();
  int stride = gridDim.x * blockDim.x;
  for (int p = blockIdx.x * blockDim.x + threadIdx.x; p < P; p += stride) {
    int i = idx[p];
    int j = idx[P + p];
    float cov = s_cr[species[i]] + s_cr[species[j]];
    float d = dist[p];
    // counting = 1 / (1 + exp(-K1*(K2*cov/d - 1)))
    //          = 1 / (1 + exp2(NEGK1L2E*(K2*cov/d) - NEGK1L2E))
    float u = K2c * cov * fast_rcp(d);
    float e2 = fast_exp2(__builtin_fmaf(NEGK1L2E, u, -NEGK1L2E));
    float counting = fast_rcp(1.0f + e2);
    atomic_fadd(&cn[i], counting);
    atomic_fadd(&cn[j], counting);
  }
}

// ---------------- pass 2: C6 interpolation (WMMA) + BJ damping + reduce ----------------
__global__ __launch_bounds__(256)
void disp_pass_kernel(const int* __restrict__ idx, const int* __restrict__ species,
                      const float* __restrict__ dist, const float* __restrict__ sqrtq,
                      const float* __restrict__ c6ref, const float* __restrict__ cna,
                      const float* __restrict__ cnb, const float* __restrict__ cn,
                      float* __restrict__ acc, int P) {
  // interleaved tables: one float4 {a, b, c6, 0} per (combo, k), padded stride 32
  __shared__ float4 s_abc[NCOMBO * TABSTR];
  __shared__ float  s_sq[NELEM];
  for (int t = threadIdx.x; t < NCOMBO * TABSTR; t += blockDim.x) {
    int combo = t >> 5;
    int k     = t & (TABSTR - 1);
    float4 v = {0.0f, 0.0f, 0.0f, 0.0f};
    if (k < NREF2) {
      int src = combo * NREF2 + k;
      v.x = cna[src];
      v.y = cnb[src];
      v.z = c6ref[src];
    }
    s_abc[t] = v;
  }
  if (threadIdx.x < NELEM) s_sq[threadIdx.x] = sqrtq[threadIdx.x];
  __syncthreads();

  const int lane = threadIdx.x & 31;
  const int half = lane >> 4;      // 0: K 0-7,16-23  1: K 8-15,24-31
  const int m    = lane & 15;      // pair slot within group; also D column N

  // post-WMMA row distribution: this lane handles row r, D element mm
  const int  mm = m & 7;
  const int  r  = 8 * half + mm;
  const bool b0 = (mm & 1) != 0, b1 = (mm & 2) != 0, b2 = (mm & 4) != 0;
  const float contrib = (m < 8) ? 1.0f : 0.0f;   // count each row exactly once

  v16h ones;
#pragma unroll
  for (int e = 0; e < 16; ++e) ones[e] = (_Float16)1.0f;
  const v8f zero8 = {};

  // wave-uniform loop bounds in SGPRs -> scalar loop control, EXEC stays all-1s
  const int wid = __builtin_amdgcn_readfirstlane(
      (int)((blockIdx.x * blockDim.x + threadIdx.x) >> 5));
  const int nw  = (int)((gridDim.x * blockDim.x) >> 5);
  const int G   = (P + 15) >> 4;   // groups of 16 pairs

  float lacc = 0.0f;
  for (int g = wid; g < G; g += nw) {
    int p = (g << 4) + m;
    bool valid = (p < P);
    int pc = valid ? p : 0;
    int i = idx[pc];
    int j = idx[P + pc];
    int si = species[i];
    int sj = species[j];
    int base = (si * NELEM + sj) << 5;     // combo * TABSTR
    float cni = cn[i];
    float cnj = cn[j];
    float d = dist[pc];
    float sqab = s_sq[si] * s_sq[sj];
    float validf = valid ? 1.0f : 0.0f;

    // Build A fragments per ISA 16-bit A-matrix 16x32 layout:
    // lane half0 holds K = e (e<8), e+8 (e>=8); half1 holds K = e+8 (e<8), e+16 (e>=8)
    v16h ag, az;
#pragma unroll
    for (int e = 0; e < 16; ++e) {
      int k = e + 8 * half + (e & 8);      // < 32, always within padded stride
      float4 abc = s_abc[base + k];        // one ds_load_b128
      float dx = cni - abc.x;
      float dy = cnj - abc.y;
      float t  = dx * dx;
      t = __builtin_fmaf(dy, dy, t);
      float gg = fast_exp2(NEGK3L2E * t);
      float gz = gg * abc.z;
      bool kv = (k < NREF2);               // uniform per lane (cndmask, no branch)
      ag[e] = (_Float16)(kv ? gg : 0.0f);
      az[e] = (_Float16)(kv ? gz : 0.0f);
    }

    // Row sums via WMMA with all-ones B (layout independent):
    // D[m,n] = sum_k A[m,k]  -- identical across columns n.
    v8f accw = __builtin_amdgcn_wmma_f32_16x16x32_f16(
        false, ag, false, ones, (short)0, zero8, false, false);
    v8f accz = __builtin_amdgcn_wmma_f32_16x16x32_f16(
        false, az, false, ones, (short)0, zero8, false, false);

    // D layout: half0 lanes hold rows 0-7 (element v -> M=v); half1 lanes rows 8-15.
    // Each lane processes ONE row (r) using element mm of its D fragments.
    float dr  = __shfl(d, r, 32);          // pair r data lives in lane r (and r+16)
    float sqr = __shfl(sqab, r, 32);
    float vr  = __shfl(validf, r, 32);
    float w = sel8(accw, b0, b1, b2);
    float z = sel8(accz, b0, b1, b2);
    float c6 = z * fast_rcp(w);
    float c8 = 3.0f * c6 * sqr;
    float bj = A1c * fast_sqrt(3.0f * sqr) + A2c;
    float d2 = dr * dr, d6 = d2 * d2 * d2, d8 = d6 * d2;
    float bj2 = bj * bj, bj6 = bj2 * bj2 * bj2, bj8 = bj6 * bj2;
    float tb = (S6c * c6) * fast_rcp(d6 + bj6) + (S8c * c8) * fast_rcp(d8 + bj8);
    lacc += contrib * vr * tb;
  }

  // wave32 reduction, one atomic per wave
#pragma unroll
  for (int off = 16; off > 0; off >>= 1)
    lacc += __shfl_xor(lacc, off, 32);
  if (lane == 0) atomic_fadd(acc, lacc);
}

// ---------------- finalize ----------------
__global__ void finalize_kernel(const float* __restrict__ energies,
                                const float* __restrict__ acc,
                                float* __restrict__ out) {
  if (threadIdx.x == 0 && blockIdx.x == 0) out[0] = energies[0] - acc[0];
}

extern "C" void kernel_launch(void* const* d_in, const int* in_sizes, int n_in,
                              void* d_out, int out_size, void* d_ws, size_t ws_size,
                              hipStream_t stream) {
  const int*   species  = (const int*)d_in[0];
  const float* energies = (const float*)d_in[1];
  const int*   idx      = (const int*)d_in[2];
  const float* dist     = (const float*)d_in[3];
  const float* cr       = (const float*)d_in[4];
  const float* sqrtq    = (const float*)d_in[5];
  const float* c6ref    = (const float*)d_in[6];
  const float* cna      = (const float*)d_in[7];
  const float* cnb      = (const float*)d_in[8];

  const int NA = in_sizes[0];       // 100000 atoms
  const int P  = in_sizes[3];       // 4000000 pairs

  float* cn  = (float*)d_ws;        // NA floats
  float* acc = cn + NA;             // 1 float accumulator

  int nz = NA + 1;
  zero_ws_kernel<<<(nz + 255) / 256, 256, 0, stream>>>(cn, nz);
  cn_pass_kernel<<<2048, 256, 0, stream>>>(idx, species, dist, cr, cn, P);
  disp_pass_kernel<<<4096, 256, 0, stream>>>(idx, species, dist, sqrtq,
                                             c6ref, cna, cnb, cn, acc, P);
  finalize_kernel<<<1, 1, 0, stream>>>(energies, acc, (float*)d_out);
}
